// SelfAttention_65721589563970
// MI455X (gfx1250) — matrix operand order repaired and checked by reading, not compiled, not run on hardware
//
#include <hip/hip_runtime.h>

#define DI __device__ __forceinline__

typedef __attribute__((ext_vector_type(16))) __bf16 v16bf;
typedef __attribute__((ext_vector_type(8)))  __bf16 bf16x8;
typedef __attribute__((ext_vector_type(8)))  float  v8f;
typedef __attribute__((ext_vector_type(4)))  float  f32x4;

constexpr int Bb = 8;
constexpr int Tt = 2048;
constexpr int Cc = 1024;
constexpr int Hh = 256;

DI unsigned short bfbits(float f) {
  return __builtin_bit_cast(unsigned short, (__bf16)f);   // native RTNE cvt
}

DI v8f wmma_bf16(v16bf a, v16bf b, v8f c) {
  return __builtin_amdgcn_wmma_f32_16x16x32_bf16(false, a, false, b, (short)0, c,
                                                 false, false);
}

// ---------------------------------------------------------------------------
// A-fragment layout (16x32 bf16 tile):  flat = (chunkBase + lane)*16 + e
//   lane = (row&15) + 16*(j>>4),  e = j&15  (contiguous per lane)
// B-fragment layout (32x16 bf16 tile):  lane n: col n; lanes<16: K 0..15,
//   lanes>=16: K 16..31, element e = K offset (contiguous per lane)
// ---------------------------------------------------------------------------

// Prepass A: x (fp32 [B*T][C]) -> xswz (bf16, A-frag layout).
__global__ __launch_bounds__(256)
void xswz_kernel(const float* __restrict__ x, unsigned short* __restrict__ xswz)
{
  const unsigned tid  = blockIdx.x * 256 + threadIdx.x;  // (mt,kc,lane,half)
  const int half = tid & 1;
  const int lane = (tid >> 1) & 31;
  const int kc   = (tid >> 6) & 31;
  const int mt   = tid >> 11;                             // 0..1023

  const int row   = mt * 16 + (lane & 15);
  const int abase = (lane >= 16) ? 16 : 0;
  const int col0  = kc * 32 + abase + half * 8;

  const float* src = x + (size_t)row * Cc + col0;
  f32x4 f0 = *(const f32x4*)(src);
  f32x4 f1 = *(const f32x4*)(src + 4);

  bf16x8 v;
#pragma unroll
  for (int i = 0; i < 4; ++i) { v[i] = (__bf16)f0[i]; v[i + 4] = (__bf16)f1[i]; }

  unsigned short* dst =
      xswz + ((((size_t)mt * 32 + kc) * 32 + lane) * 16) + half * 8;
  *(bf16x8*)dst = v;
}

// Prepass B: Wq/Wk/Wv (fp32 [C][H]) -> wswz (bf16, B-frag layout,
// [mat][kc 32][nt 16][lane 32][e 16]).
__global__ __launch_bounds__(256)
void wswz_kernel(const float* __restrict__ Wq, const float* __restrict__ Wk,
                 const float* __restrict__ Wv, unsigned short* __restrict__ wswz)
{
  const unsigned t = blockIdx.x * 256 + threadIdx.x;
  const int e    = t & 15;
  const int lane = (t >> 4) & 31;
  const int nt   = (t >> 9) & 15;
  const int kc   = (t >> 13) & 31;
  const int mat  = t >> 18;                               // 0..2

  const float* W = (mat == 0) ? Wq : (mat == 1) ? Wk : Wv;
  const int krow = kc * 32 + ((lane >= 16) ? 16 : 0) + e;
  const int col  = nt * 16 + (lane & 15);
  wswz[t] = bfbits(W[(size_t)krow * Hh + col]);
}

// ---------------------------------------------------------------------------
// Kernel 1: q/k/v = x @ W{q,k,v}. Inner loop = 2 contiguous v16bf loads + wmma.
// q stored pre-swizzled (A-frag layout) and pre-scaled by sqrt(H)=16 (exact
// power-of-2 in bf16, faithful to the reference's scale bug);
// k stored [T][H]; v stored transposed [B][H][T].
// ---------------------------------------------------------------------------
__global__ __launch_bounds__(256)
void proj_kernel(const unsigned short* __restrict__ xswz,
                 const unsigned short* __restrict__ wswz,
                 unsigned short* __restrict__ qws,
                 unsigned short* __restrict__ kws,
                 unsigned short* __restrict__ vtws)
{
  const int w    = threadIdx.x >> 5;
  const int lane = threadIdx.x & 31;
  const int ln   = lane & 15;
  const int hi   = lane >> 4;

  const int mt  = blockIdx.x;                    // 16-row tile of B*T
  const int mat = blockIdx.y >> 1;               // 0=q 1=k 2=v
  const int nt  = ((blockIdx.y & 1) << 3) + w;   // 16-col tile of H

  const unsigned short* ap =
      xswz + (((size_t)mt * 32) * 32 + lane) * 16;                 // += kc*512
  const unsigned short* bp =
      wswz + ((((size_t)mat * 32) * 16 + nt) * 32 + lane) * 16;    // += kc*8192

  v8f acc = {0.f, 0.f, 0.f, 0.f, 0.f, 0.f, 0.f, 0.f};

#pragma unroll 4
  for (int kc = 0; kc < 32; ++kc) {
    v16bf a  = *(const v16bf*)(ap + (size_t)kc * 512);
    v16bf bv = *(const v16bf*)(bp + (size_t)kc * 8192);
    acc = wmma_bf16(a, bv, acc);
  }

  const int col = nt * 16 + ln;
  if (mat == 0) {
    // q in A-frag layout, pre-scaled by 16 (exact in bf16)
    const int kc    = nt >> 1;
    const int laneH = 16 * (nt & 1);
    const int e     = ln;
    unsigned short* qb = qws + (((size_t)mt * 8 + kc) * 32) * 16;
#pragma unroll
    for (int r = 0; r < 8; ++r) {
      const int laneP = (r + (hi ? 8 : 0)) + laneH;
      qb[(size_t)laneP * 16 + e] = bfbits(acc[r] * 16.0f);
    }
  } else if (mat == 1) {
#pragma unroll
    for (int r = 0; r < 8; ++r) {
      const int row = mt * 16 + r + (hi ? 8 : 0);
      kws[(size_t)row * Hh + col] = bfbits(acc[r]);
    }
  } else {
#pragma unroll
    for (int r = 0; r < 8; ++r) {
      const int row = mt * 16 + r + (hi ? 8 : 0);
      const int bb  = row >> 11;                 // / T
      const int tt  = row & (Tt - 1);
      vtws[(size_t)bb * Hh * Tt + (size_t)col * Tt + tt] = bfbits(acc[r]);
    }
  }
}

// ---------------------------------------------------------------------------
// Kernel 2: flash attention. Wave = 16 query rows x full H=256; key blocks of
// 32; online softmax. All WMMA operands are single contiguous 32B v16bf loads;
// softmax row-sum comes from an extra WMMA against an all-ones B fragment.
// ---------------------------------------------------------------------------
__global__ __launch_bounds__(256)
void attn_kernel(const unsigned short* __restrict__ qws,
                 const unsigned short* __restrict__ kws,
                 const unsigned short* __restrict__ vtws,
                 float* __restrict__ out)
{
  __shared__ __align__(16) unsigned short pLds[8][16 * 32];

  const int w    = threadIdx.x >> 5;
  const int lane = threadIdx.x & 31;
  const int ln   = lane & 15;
  const int hi   = lane >> 4;

  const int b  = blockIdx.y;
  const int t0 = blockIdx.x * 128 + w * 16;

  const int tileG = (b * Tt + t0) >> 4;
  const unsigned short* qfp = qws + (((size_t)tileG * 8) * 32 + lane) * 16;
  const unsigned short* kp  = kws  + (size_t)b * Tt * Hh;   // [T][H]
  const unsigned short* vtp = vtws + (size_t)b * Hh * Tt;   // [H][T]

  const v8f vzero = {0.f, 0.f, 0.f, 0.f, 0.f, 0.f, 0.f, 0.f};
  v8f Oacc[16];
#pragma unroll
  for (int i = 0; i < 16; ++i) Oacc[i] = vzero;
  v8f Lacc = vzero;                               // softmax denominator tile

  float mrow[8];
#pragma unroll
  for (int r = 0; r < 8; ++r) mrow[r] = -__builtin_inff();

  // all-ones B fragment (bf16 1.0 = 0x3F80) for row-sum WMMA
  v16bf bones;
#pragma unroll
  for (int i = 0; i < 16; ++i)
    bones[i] = __builtin_bit_cast(__bf16, (unsigned short)0x3F80);

  // P-store swizzle (A-frag order): per-lane constants
  const int pE0   = ln;
  const int pLanH = 0;
  unsigned short* myLds = pLds[w];

  const int nblk = (t0 + 47) >> 5;               // causal key blocks of 32
  for (int blk = 0; blk < nblk; ++blk) {
    const int j0 = blk << 5;

    // ---- S = Q @ K^T over H=256 (q pre-scaled by sqrt(H)) ----
    v8f S0 = vzero, S1 = vzero;
    const unsigned short* k0p = kp + (size_t)(j0 + ln) * Hh + (hi ? 16 : 0);
    const unsigned short* k1p = k0p + (size_t)16 * Hh;
#pragma unroll
    for (int kc = 0; kc < 8; ++kc) {
      v16bf a  = *(const v16bf*)(qfp + (size_t)kc * 512);
      v16bf b0 = *(const v16bf*)(k0p + kc * 32);
      S0 = wmma_bf16(a, b0, S0);
      v16bf b1 = *(const v16bf*)(k1p + kc * 32);
      S1 = wmma_bf16(a, b1, S1);
    }

    // ---- causal mask + online softmax (max via shfl; sum via ones-WMMA) ----
    const bool needMask = (j0 + 31 > t0);
#pragma unroll
    for (int r = 0; r < 8; ++r) {
      const int i = t0 + r + (hi ? 8 : 0);       // global query row
      float s0 = S0[r];
      float s1 = S1[r];
      if (needMask) {
        if (j0 + ln      > i) s0 = -__builtin_inff();
        if (j0 + 16 + ln > i) s1 = -__builtin_inff();
      }
      float mx = fmaxf(s0, s1);
      mx = fmaxf(mx, __shfl_xor(mx, 1));
      mx = fmaxf(mx, __shfl_xor(mx, 2));
      mx = fmaxf(mx, __shfl_xor(mx, 4));
      mx = fmaxf(mx, __shfl_xor(mx, 8));
      const float mnew  = fmaxf(mrow[r], mx);
      const float alpha = __expf(mrow[r] - mnew);
      const float p0 = __expf(s0 - mnew);
      const float p1 = __expf(s1 - mnew);
      mrow[r] = mnew;
      Lacc[r] *= alpha;
#pragma unroll
      for (int ht = 0; ht < 16; ++ht) Oacc[ht][r] *= alpha;

      // store P directly in A-frag element order
      const int laneP = (r + (hi ? 8 : 0)) + pLanH;
      myLds[laneP * 16 + pE0]     = bfbits(p0);
      myLds[(laneP + 16) * 16 + pE0] = bfbits(p1);
    }

    // per-wave LDS region; DS ops are in-order within a wave
    __asm__ volatile("s_wait_dscnt 0" ::: "memory");

    // ---- P frag (one contiguous v16bf); L += P@1 ; O += P@V ----
    v16bf pa = *(const v16bf*)(myLds + lane * 16);
    Lacc = wmma_bf16(pa, bones, Lacc);

    const unsigned short* vbase = vtp + j0 + (hi ? 16 : 0);
#pragma unroll
    for (int ht = 0; ht < 16; ++ht) {
      v16bf vb = *(const v16bf*)(vbase + (size_t)(ht * 16 + ln) * Tt);
      Oacc[ht] = wmma_bf16(pa, vb, Oacc[ht]);
    }
  }

  // ---- normalize (rcp once per row) and store fp32 output ----
  float inv[8];
#pragma unroll
  for (int r = 0; r < 8; ++r) inv[r] = __builtin_amdgcn_rcpf(Lacc[r]);
#pragma unroll
  for (int ht = 0; ht < 16; ++ht) {
#pragma unroll
    for (int r = 0; r < 8; ++r) {
      const int trow = t0 + r + (hi ? 8 : 0);
      out[((size_t)b * Tt + trow) * Hh + ht * 16 + ln] = Oacc[ht][r] * inv[r];
    }
  }
}

extern "C" void kernel_launch(void* const* d_in, const int* in_sizes, int n_in,
                              void* d_out, int out_size, void* d_ws, size_t ws_size,
                              hipStream_t stream)
{
  const float* x  = (const float*)d_in[0];
  const float* Wq = (const float*)d_in[1];
  const float* Wk = (const float*)d_in[2];
  const float* Wv = (const float*)d_in[3];
  float* out = (float*)d_out;

  // workspace (bf16/ushort): xswz 32MB | wswz 1.5MB | q 8MB | k 8MB | vT 8MB
  unsigned short* xswz = (unsigned short*)d_ws;
  unsigned short* wswz = xswz + (size_t)Bb * Tt * Cc;
  unsigned short* qws  = wswz + (size_t)3 * 32 * 16 * 32 * 16;
  unsigned short* kws  = qws + (size_t)Bb * Tt * Hh;
  unsigned short* vtws = kws + (size_t)Bb * Tt * Hh;

  // prepass: swizzle x and W into exact WMMA fragment layouts (bf16)
  xswz_kernel<<<(Bb * Tt / 16) * 32 * 32 * 2 / 256, 256, 0, stream>>>(x, xswz);
  wswz_kernel<<<3 * 32 * 16 * 32 * 16 / 256, 256, 0, stream>>>(Wq, Wk, Wv, wswz);

  dim3 g1((Bb * Tt) / 16, 6);          // 1024 row-tiles x {q,k,v}x{col half}
  proj_kernel<<<g1, 256, 0, stream>>>(xswz, wswz, qws, kws, vtws);

  dim3 g2(Tt / 128, Bb);               // 16 query blocks x 8 batches
  attn_kernel<<<g2, 256, 0, stream>>>(qws, kws, vtws, out);
}
